// RHWF_66151086293064
// MI455X (gfx1250) — compile-verified, matrix operand order checked by reading, not run on hardware
//
#include <hip/hip_runtime.h>

typedef __attribute__((ext_vector_type(16))) __bf16 bf16x16;
typedef __attribute__((ext_vector_type(8)))  float  f32x8;
typedef __attribute__((ext_vector_type(4)))  int    i32x4;

#if defined(__has_builtin)
#if __has_builtin(__builtin_amdgcn_global_load_async_to_lds_b128) && \
    __has_builtin(__builtin_amdgcn_s_wait_asynccnt)
#define USE_ASYNC_LDS 1
#endif
#endif

#ifdef USE_ASYNC_LDS
// builtin signature (from probe diagnostic): param0 = v4i addrspace(1)*, param1 = LDS side
#define GLBP(p) ((__attribute__((address_space(1))) i32x4*)(p))
#define LDSP(p) ((__attribute__((address_space(3))) i32x4*)(p))
#endif

static __device__ __forceinline__ unsigned short f2bf(float x){
  union { float f; unsigned u; } v; v.f = x;
  unsigned u = v.u;
  u += 0x7FFFu + ((u >> 16) & 1u);           // round-to-nearest-even
  return (unsigned short)(u >> 16);
}

// ---- workspace byte offsets ------------------------------------------------
#define OFF_F2T  0ul                                   // fmap2 bf16 [b][hw][96]
#define OFF_X1   (OFF_F2T + 128ul*1024ul*96ul*2ul)     // corr_cat bf16 [row][160]
#define OFF_CF   (OFF_X1  + 128ul*1024ul*160ul*2ul)    // cf bf16 [row][160]
#define OFF_W1T  (OFF_CF  + 128ul*1024ul*160ul*2ul)    // w1 bf16 [o][160]
#define OFF_W3T  (OFF_W1T + 128ul*160ul*2ul)           // w3 bf16 [tap][o][160]
#define OFF_FLOW (OFF_W3T + 9ul*128ul*160ul*2ul)       // flow f32 (B,2,1024)
#define OFF_D4   (OFF_FLOW + 128ul*2ul*1024ul*4ul)     // delta4 f32 (64)

// ---- per-batch DLT homography (8x8 Gauss w/ partial pivot) -----------------
static __device__ void dlt_solve(const float* fp, float* H){
  const float xs[4] = {0.f, 31.f, 0.f, 31.f};
  const float ys[4] = {0.f, 0.f, 31.f, 31.f};
  float M[8][9];
  for (int p = 0; p < 4; ++p){
    float x = xs[p], y = ys[p];
    float u = x + fp[p]     * 0.25f;   // four_point / A_SCALE
    float v = y + fp[4 + p] * 0.25f;
    float r0[9] = {x, y, 1.f, 0.f, 0.f, 0.f, -x*u, -y*u, u};
    float r1[9] = {0.f, 0.f, 0.f, x, y, 1.f, -x*v, -y*v, v};
    for (int j = 0; j < 9; ++j){ M[p][j] = r0[j]; M[p+4][j] = r1[j]; }
  }
  for (int c = 0; c < 8; ++c){
    int piv = c; float best = fabsf(M[c][c]);
    for (int r = c+1; r < 8; ++r){ float a = fabsf(M[r][c]); if (a > best){ best = a; piv = r; } }
    if (piv != c) for (int j = 0; j < 9; ++j){ float tmp = M[c][j]; M[c][j] = M[piv][j]; M[piv][j] = tmp; }
    float inv = 1.f / M[c][c];
    for (int j = c; j < 9; ++j) M[c][j] *= inv;
    for (int r = 0; r < 8; ++r){
      if (r == c) continue;
      float f = M[r][c];
      for (int j = c; j < 9; ++j) M[r][j] -= f * M[c][j];
    }
  }
  for (int c = 0; c < 8; ++c) H[c] = M[c][8];
  H[8] = 1.f;
}

// ---- weight prep: bf16 transposed weights + zero delta4 --------------------
__global__ __launch_bounds__(256) void k_prep_w(const float* w1, const float* w3, float* d4,
                                                unsigned short* w1t, unsigned short* w3t){
  int t = blockIdx.x * 256 + threadIdx.x;
  if (t < 64) d4[t] = 0.f;
  if (t < 20480){                       // W1T[o][k], K padded 145->160
    int o = t / 160, k = t % 160;
    w1t[t] = f2bf(k < 145 ? w1[o*145 + k] : 0.f);
  } else if (t < 204800){               // W3T[tap][o][k], K padded 130->160
    int i = t - 20480;
    int tap = i / (128*160);
    int rem = i % (128*160);
    int o = rem / 160, k = rem % 160;
    float v = 0.f;
    if (k < 130) v = w3[((o*130 + k)*3 + tap/3)*3 + (tap%3)];
    w3t[i] = f2bf(v);
  }
}

// ---- fmap2 f32 NCHW -> bf16 [b][hw][c] -------------------------------------
__global__ __launch_bounds__(256) void k_prep_f2(const float* fmap2, unsigned short* f2t){
  int b = blockIdx.x, t = threadIdx.x;
  for (int j = 0; j < 4; ++j){
    int p = t + j * 256;
    size_t obase = ((size_t)b * 1024 + p) * 96;
    for (int c = 0; c < 96; ++c)
      f2t[obase + c] = f2bf(fmap2[((size_t)(b*96 + c)) * 1024 + p]);
  }
}

// ---- initial flow = warp(coords) - coords0 ---------------------------------
__global__ __launch_bounds__(256) void k_flow_init(const float* four_point, float* flow){
  __shared__ float Hs[9];
  int b = blockIdx.x, t = threadIdx.x;
  if (t == 0){
    float fp[8];
    for (int i = 0; i < 8; ++i) fp[i] = four_point[b*8 + i];
    dlt_solve(fp, Hs);
  }
  __syncthreads();
  for (int j = 0; j < 4; ++j){
    int n = t + j*256;
    int i = n >> 5, x = n & 31;
    float X = (float)x, Y = (float)i;
    float d = Hs[6]*X + Hs[7]*Y + Hs[8];
    float inv = 1.f / d;
    flow[((size_t)b*2  )*1024 + n] = (Hs[0]*X + Hs[1]*Y + Hs[2]) * inv - X;
    flow[((size_t)b*2+1)*1024 + n] = (Hs[3]*X + Hs[4]*Y + Hs[5]) * inv - Y;
  }
}

// ---- correlation via tile GEMM + relu + pool/crop -> X1 --------------------
// block = (h,b): two 16-pixel M-tiles x 34 N-tiles over a 17x32 f2 region.
__global__ __launch_bounds__(256) void k_corr(const float* fmap1, const unsigned short* f2t,
                                              unsigned short* x1){
  __shared__ alignas(32) unsigned short As[32][96];     // f1 row, bf16
  __shared__ alignas(32) float Corr[2][16][544];        // 16 x (17*32) per tile
  int h = blockIdx.x, b = blockIdx.y;
  int t = threadIdx.x, wv = t >> 5, lane = t & 31;
  int lm = lane & 15, kh = lane >> 4;

  for (int idx = t; idx < 3072; idx += 256){            // 32 px * 96 ch
    int c = idx >> 5, px = idx & 31;
    As[px][c] = f2bf(fmap1[(((size_t)b*96 + c) << 10) + (h << 5) + px]);
  }
  __syncthreads();

  const bf16x16 bz = {};
  for (int job = wv; job < 68; job += 8){               // 2 Mtiles * 34 Ntiles
    int mt = job & 1, nt = job >> 1;
    int x0 = mt << 4;
    int cn = (nt << 4) + lm;                            // region column 0..543
    int dy = cn >> 5, xx = cn & 31;
    int gy = h - 8 + dy, gx = x0 - 8 + xx;
    bool inb = (gy >= 0) && (gy < 32) && (gx >= 0) && (gx < 32);
    int cy = inb ? gy : 0, cx = inb ? gx : 0;
    const unsigned short* bp = f2t + (((size_t)b*1024 + cy*32 + cx) * 96);
    f32x8 acc = {};
#pragma unroll
    for (int ks = 0; ks < 3; ++ks){                     // K = 96 = 3 x 32
      int ko = ks*32 + kh*16;
      bf16x16 a  = *(const bf16x16*)&As[x0 + lm][ko];
      bf16x16 bm = inb ? *(const bf16x16*)(bp + ko) : bz;
      acc = __builtin_amdgcn_wmma_f32_16x16x32_bf16(false, a, false, bm,
                                                    (short)0, acc, false, false);
    }
#pragma unroll
    for (int v = 0; v < 8; ++v)                          // C layout: M=v+8*kh, N=lm
      Corr[mt][v + kh*8][cn] = acc[v];
  }
  __syncthreads();

  // relu, 2x2 avgpool over [:16,:16] (64 ch) + 9x9 center crop (81 ch) -> 160
  for (int idx = t; idx < 5120; idx += 256){
    int x = idx & 31, ch = idx >> 5;
    int mt = x >> 4, m = x & 15;
    float val = 0.f;
    if (ch < 64){
      int dy0 = (ch >> 3) * 2, dx0 = (ch & 7) * 2;
      float s = 0.f;
#pragma unroll
      for (int a2 = 0; a2 < 2; ++a2)
#pragma unroll
        for (int b2 = 0; b2 < 2; ++b2){
          float c5 = Corr[mt][m][(dy0 + a2)*32 + m + dx0 + b2];
          s += c5 > 0.f ? c5 : 0.f;
        }
      val = s * 0.25f;
    } else if (ch < 145){
      int v2 = ch - 64;
      int vy = v2 / 9 + 4, vx = v2 % 9 + 4;
      float c5 = Corr[mt][m][vy*32 + m + vx];
      val = c5 > 0.f ? c5 : 0.f;
    }
    x1[(((size_t)b*1024) + (h << 5) + x)*160 + ch] = f2bf(val);
  }
}

// ---- conv1 (1x1, 145->128) GEMM + assemble cf (c1 | flow | pad) ------------
__global__ __launch_bounds__(256) void k_conv1(const unsigned short* x1, const unsigned short* w1t,
                                               const float* b1, const float* flow,
                                               unsigned short* cf){
  __shared__ alignas(32) unsigned short Arows[64*160];
  int t = threadIdx.x, wv = t >> 5, lane = t & 31;
  int lm = lane & 15, kh = lane >> 4;
  size_t rowbase = (size_t)blockIdx.x * 64;

  const unsigned int* xu = (const unsigned int*)x1;
  unsigned int* au = (unsigned int*)Arows;
#ifdef USE_ASYNC_LDS
  // 64 rows * 320B = 1280 x b128 async global->LDS transfers (ASYNCcnt path)
  for (int idx = t; idx < 1280; idx += 256)
    __builtin_amdgcn_global_load_async_to_lds_b128(
        GLBP(xu + rowbase*80 + idx*4), LDSP(au + idx*4), 0, 0);
  __builtin_amdgcn_s_wait_asynccnt(0);
  __syncthreads();
#else
  for (int idx = t; idx < 5120; idx += 256)             // 64 rows * 80 dwords
    au[idx] = xu[rowbase*80 + idx];
  __syncthreads();
#endif

  for (int tile = wv; tile < 32; tile += 8){            // 4 Mtiles x 8 Ntiles
    int mt = tile >> 3, nt = tile & 7;
    f32x8 acc = {};
    const unsigned short* bcol = w1t + ((nt << 4) + lm) * 160;
#pragma unroll
    for (int ks = 0; ks < 5; ++ks){                     // K = 160
      int ko = ks*32 + kh*16;
      bf16x16 a  = *(const bf16x16*)&Arows[((mt << 4) + lm)*160 + ko];
      bf16x16 bm = *(const bf16x16*)(bcol + ko);
      acc = __builtin_amdgcn_wmma_f32_16x16x32_bf16(false, a, false, bm,
                                                    (short)0, acc, false, false);
    }
    int o = (nt << 4) + lm;
    float bias = b1[o];
#pragma unroll
    for (int v = 0; v < 8; ++v){
      int r = (mt << 4) + v + kh*8;
      float val = acc[v] + bias;
      val = val > 0.f ? val : 0.f;
      cf[(rowbase + r)*160 + o] = f2bf(val);
    }
  }

  // channels 128/129 = flow, 130..159 = zero pad
  for (int idx = t; idx < 2048; idx += 256){
    int r = idx >> 5, ch = 128 + (idx & 31);
    size_t row = rowbase + r;
    int b = (int)(row >> 10), hw = (int)(row & 1023);
    float val = 0.f;
    if (ch == 128)      val = flow[((size_t)b*2  )*1024 + hw];
    else if (ch == 129) val = flow[((size_t)b*2+1)*1024 + hw];
    cf[row*160 + ch] = f2bf(val);
  }
}

// ---- conv3 (3x3, 130->128) implicit GEMM + fused head pooling --------------
__global__ __launch_bounds__(256) void k_conv3(const unsigned short* cf, const unsigned short* w3t,
                                               const float* b3, const float* wh, const float* bh,
                                               float* d4){
  __shared__ alignas(32) unsigned short As[3*34*160];   // 3-row halo, 34 px each
  __shared__ float C3[32][128];
  int h = blockIdx.x, b = blockIdx.y;
  int t = threadIdx.x, wv = t >> 5, lane = t & 31;
  int lm = lane & 15, kh = lane >> 4;

  const unsigned int* cfu = (const unsigned int*)cf;
  unsigned int* au = (unsigned int*)As;
#ifdef USE_ASYNC_LDS
  // interior 32 pixels of each in-bounds halo row: async b128; rest zero-filled
  for (int yy = 0; yy < 3; ++yy){
    int gy = h - 1 + yy;
    unsigned int* dst = au + (yy*34 + 1)*80;            // xx = 1..32
    if (gy >= 0 && gy < 32){
      const unsigned int* src = cfu + (((size_t)b*1024) + gy*32)*80;
      for (int idx = t; idx < 640; idx += 256)          // 32 px * 320B / 16B
        __builtin_amdgcn_global_load_async_to_lds_b128(
            GLBP(src + idx*4), LDSP(dst + idx*4), 0, 0);
    } else {
      for (int idx = t; idx < 2560; idx += 256) dst[idx] = 0u;
    }
  }
  for (int idx = t; idx < 480; idx += 256){             // border cols x=-1, x=32
    int yy = idx / 160, rem = idx % 160;
    int xx = (rem < 80) ? 0 : 33;
    au[(yy*34 + xx)*80 + (rem % 80)] = 0u;
  }
  __builtin_amdgcn_s_wait_asynccnt(0);
  __syncthreads();
#else
  for (int idx = t; idx < 8160; idx += 256){            // 102 px * 80 dwords
    int pix = idx / 80, c32 = idx % 80;
    int yy = pix / 34, xx = pix % 34;
    int gy = h - 1 + yy, gx = xx - 1;
    unsigned int v = 0u;
    if (gy >= 0 && gy < 32 && gx >= 0 && gx < 32)
      v = cfu[(((size_t)b*1024) + gy*32 + gx)*80 + c32];
    au[idx] = v;
  }
  __syncthreads();
#endif

  for (int tile = wv; tile < 16; tile += 8){            // 2 Mtiles x 8 Ntiles
    int mt = tile >> 3, nt = tile & 7;
    int o = (nt << 4) + lm;
#pragma unroll
    for (int tap = 0; tap < 9; ++tap)                   // warm L2 for tap columns
      __builtin_prefetch(w3t + ((size_t)(tap*128 + o))*160, 0, 1);
    f32x8 acc = {};
#pragma unroll
    for (int tap = 0; tap < 9; ++tap){
      int dy = tap / 3, dx = tap % 3;
      const unsigned short* arow = &As[((dy*34) + (mt << 4) + lm + dx)*160];
      const unsigned short* bcol = w3t + ((size_t)(tap*128 + o))*160;
#pragma unroll
      for (int ks = 0; ks < 5; ++ks){                   // 9 taps * 5 ksteps = 45 WMMAs
        int ko = ks*32 + kh*16;
        bf16x16 a  = *(const bf16x16*)(arow + ko);
        bf16x16 bm = *(const bf16x16*)(bcol + ko);
        acc = __builtin_amdgcn_wmma_f32_16x16x32_bf16(false, a, false, bm,
                                                      (short)0, acc, false, false);
      }
    }
    float bias = b3[o];
#pragma unroll
    for (int v = 0; v < 8; ++v){
      float val = acc[v] + bias;
      C3[(mt << 4) + v + kh*8][o] = val > 0.f ? val : 0.f;
    }
  }
  __syncthreads();

  // head: delta = wh*c3 + bh, quadrant-mean (/256) accumulated atomically
  if (t < 64){
    int pos = t >> 1, ch = t & 1;
    float s = 0.f;
    for (int c = 0; c < 128; ++c) s += wh[ch*128 + c] * C3[pos][c];
    s += bh[ch];
    s *= (1.f / 256.f);
    atomicAdd(&d4[((b*2 + ch)*2 + (h >> 4))*2 + (pos >> 4)], s);
  }
}

// ---- final: four_point + delta4 -> homography -> coords1_new ---------------
__global__ __launch_bounds__(256) void k_final(const float* four_point, const float* d4, float* out){
  __shared__ float Hs[9];
  int b = blockIdx.x, t = threadIdx.x;
  if (t == 0){
    float fp[8];
    for (int i = 0; i < 8; ++i) fp[i] = four_point[b*8 + i] + d4[b*8 + i];
    dlt_solve(fp, Hs);
  }
  __syncthreads();
  for (int j = 0; j < 4; ++j){
    int n = t + j*256;
    int i = n >> 5, x = n & 31;
    float X = (float)x, Y = (float)i;
    float d = Hs[6]*X + Hs[7]*Y + Hs[8];
    float inv = 1.f / d;
    out[((size_t)b*2  )*1024 + n] = (Hs[0]*X + Hs[1]*Y + Hs[2]) * inv;
    out[((size_t)b*2+1)*1024 + n] = (Hs[3]*X + Hs[4]*Y + Hs[5]) * inv;
  }
}

extern "C" void kernel_launch(void* const* d_in, const int* in_sizes, int n_in,
                              void* d_out, int out_size, void* d_ws, size_t ws_size,
                              hipStream_t stream){
  (void)in_sizes; (void)n_in; (void)out_size; (void)ws_size;
  const float* fmap1      = (const float*)d_in[0];
  const float* fmap2      = (const float*)d_in[1];
  const float* four_point = (const float*)d_in[2];
  const float* w1         = (const float*)d_in[3];
  const float* b1         = (const float*)d_in[4];
  const float* w3         = (const float*)d_in[5];
  const float* b3         = (const float*)d_in[6];
  const float* wh         = (const float*)d_in[7];
  const float* bh         = (const float*)d_in[8];

  char* ws = (char*)d_ws;
  unsigned short* f2t = (unsigned short*)(ws + OFF_F2T);
  unsigned short* x1  = (unsigned short*)(ws + OFF_X1);
  unsigned short* cf  = (unsigned short*)(ws + OFF_CF);
  unsigned short* w1t = (unsigned short*)(ws + OFF_W1T);
  unsigned short* w3t = (unsigned short*)(ws + OFF_W3T);
  float* flow = (float*)(ws + OFF_FLOW);
  float* d4   = (float*)(ws + OFF_D4);
  float* out  = (float*)d_out;

  k_prep_w   <<<800, 256, 0, stream>>>(w1, w3, d4, w1t, w3t);
  k_prep_f2  <<<128, 256, 0, stream>>>(fmap2, f2t);
  k_flow_init<<<128, 256, 0, stream>>>(four_point, flow);
  k_corr     <<<dim3(32,128), 256, 0, stream>>>(fmap1, f2t, x1);
  k_conv1    <<<2048, 256, 0, stream>>>(x1, w1t, b1, flow, cf);
  k_conv3    <<<dim3(32,128), 256, 0, stream>>>(cf, w3t, b3, wh, bh, d4);
  k_final    <<<128, 256, 0, stream>>>(four_point, d4, out);
}
